// MultiHeadAttention_33981781246254
// MI455X (gfx1250) — matrix operand order repaired and checked by reading, not compiled, not run on hardware
//
#include <hip/hip_runtime.h>
#include <hip/hip_bf16.h>

// ---------------------------------------------------------------------------
// Problem constants
// ---------------------------------------------------------------------------
#define BATCH 4
#define SEQ   1024
#define DMODEL 1024
#define HEADS 16
#define DHEAD 64
#define NEG_BIG (-4294967295.0f)
#define LN_EPS 1e-3f

typedef __attribute__((ext_vector_type(16))) __bf16 v16bf;
typedef __attribute__((ext_vector_type(8)))  float  v8f;
typedef __attribute__((ext_vector_type(8)))  unsigned short u16x8;
typedef __attribute__((ext_vector_type(4)))  int    v4i;

#if __has_builtin(__builtin_amdgcn_global_load_async_to_lds_b128)
#define HAVE_ASYNC 1
typedef __attribute__((address_space(1))) v4i gv4i_t;   // global int4
typedef __attribute__((address_space(3))) v4i lv4i_t;   // LDS int4
#endif

// ---------------------------------------------------------------------------
// bf16 helpers: native hardware convert (RNE) + bit reinterpret
// ---------------------------------------------------------------------------
static __device__ __forceinline__ unsigned short bf16_bits(float f) {
    union { __bf16 b; unsigned short u; } c; c.b = (__bf16)f; return c.u;
}

union Frag16 {
    v16bf v;
    u16x8 h[2];
    __bf16 b[16];
};

// Load a 16-halfword fragment from a 32-byte-aligned contiguous run.
static __device__ __forceinline__ v16bf load_frag16(const unsigned short* p) {
    Frag16 f;
    f.h[0] = *(const u16x8*)(p);
    f.h[1] = *(const u16x8*)(p + 8);
    return f.v;
}

static __device__ __forceinline__ v8f wmma_bf16(v16bf a, v16bf b, v8f c) {
    // (neg_a, A, neg_b, B, c_mod, C, reuse_a, reuse_b)
    return __builtin_amdgcn_wmma_f32_16x16x32_bf16(false, a, false, b, (short)0, c,
                                                   false, false);
}

static __device__ __forceinline__ void wait_dscnt0() {
    asm volatile("s_wait_dscnt 0x0" ::: "memory");
}

// ===========================================================================
// Kernel 1: pack weights into bf16 B-fragment-friendly layout
//   Wpack[kt][n][kk] = bf16( W[kt*32+kk][n] )
// ===========================================================================
__global__ __launch_bounds__(256) void pack_w_kernel(
        const float* __restrict__ Wq, const float* __restrict__ Wk,
        const float* __restrict__ Wv, const float* __restrict__ Wo,
        unsigned short* __restrict__ out) {
    int z = blockIdx.y;
    const float* W = (z == 0) ? Wq : (z == 1) ? Wk : (z == 2) ? Wv : Wo;
    unsigned short* o = out + (size_t)z * (32u * 1024u * 32u);
    int idx = blockIdx.x * 256 + threadIdx.x;       // 0 .. 1048575
    int k = idx >> 10;
    int n = idx & 1023;
    o[((size_t)(k >> 5) * 1024 + n) * 32 + (k & 31)] =
        bf16_bits(W[(size_t)k * DMODEL + n]);
}

// ===========================================================================
// Kernel 2: padding masks.  mask[row] = sign(|sum_d x[row,d]|) (0 or 1).
// ===========================================================================
__global__ __launch_bounds__(256) void masks_kernel(
        const float* __restrict__ q, const float* __restrict__ k,
        float* __restrict__ qmask, float* __restrict__ kmask) {
    int wid  = (blockIdx.x * blockDim.x + threadIdx.x) >> 5;
    int lane = threadIdx.x & 31;
    if (wid >= 2 * BATCH * SEQ) return;
    const float* src = (wid < BATCH * SEQ) ? q : k;
    int row = wid & (BATCH * SEQ - 1);
    float s = 0.f;
    #pragma unroll
    for (int e = 0; e < 32; ++e) s += src[(size_t)row * DMODEL + e * 32 + lane];
    #pragma unroll
    for (int o = 16; o; o >>= 1) s += __shfl_xor(s, o, 32);
    if (lane == 0) {
        float m = (s != 0.f) ? 1.f : 0.f;
        ((wid < BATCH * SEQ) ? qmask : kmask)[row] = m;
    }
}

// ===========================================================================
// Kernel 3: QKV projection GEMM  (X[4096x1024] @ W[1024x1024] + b) -> bf16
// grid.y selects Q/K/V.  One wave computes a 16x64 tile: 4 independent WMMA
// accumulator chains (fills XDL hazard slots), A fragment + its f32->bf16
// conversions amortized over 4 WMMAs.
// V is stored TRANSPOSED per head: Vt[(b*H+h)*DH + dh][s].
// ===========================================================================
__global__ __launch_bounds__(256) void qkv_gemm_kernel(
        const float* __restrict__ Xq, const float* __restrict__ Xk,
        const float* __restrict__ Xv, const unsigned short* __restrict__ Wpack,
        const float* __restrict__ bq, const float* __restrict__ bk,
        const float* __restrict__ bv,
        unsigned short* __restrict__ Qb, unsigned short* __restrict__ Kb,
        unsigned short* __restrict__ Vt) {
    int z = blockIdx.y;
    const float* X = (z == 0) ? Xq : (z == 1) ? Xk : Xv;
    const unsigned short* W = Wpack + (size_t)z * (32u * 1024u * 32u);
    const float* bias = (z == 0) ? bq : (z == 1) ? bk : bv;
    unsigned short* OUT = (z == 0) ? Qb : (z == 1) ? Kb : Vt;

    int w = threadIdx.x >> 5, lane = threadIdx.x & 31;
    int half = lane >> 4, l16 = lane & 15;
    int tile = blockIdx.x * 8 + w;          // 0 .. 4095
    int rt = tile >> 4;                     // 256 row tiles (16 rows each)
    int cg = tile & 15;                     // 16 col groups (64 cols each)

    const float* xrow = X + (size_t)(rt * 16 + l16) * DMODEL;
    v8f acc[4];
    #pragma unroll
    for (int nt = 0; nt < 4; ++nt)
        acc[nt] = (v8f){0.f, 0.f, 0.f, 0.f, 0.f, 0.f, 0.f, 0.f};

    for (int kt = 0; kt < 32; ++kt) {
        // A fragment: lane row = l16; two contiguous 8-float runs, native cvt.
        int k0 = kt * 32 + (half ? 16 : 0);
        Frag16 a;
        #pragma unroll
        for (int e = 0; e < 8; ++e) {
            a.b[e]     = (__bf16)xrow[k0 + e];
            a.b[8 + e] = (__bf16)xrow[k0 + 8 + e];
        }
        if (kt + 1 < 32)
            __builtin_prefetch(
                W + ((size_t)(kt + 1) * 1024 + cg * 64 + l16) * 32, 0, 1);
        // 4 B fragments / 4 independent WMMA chains.
        #pragma unroll
        for (int nt = 0; nt < 4; ++nt) {
            const unsigned short* wp =
                W + ((size_t)kt * 1024 + cg * 64 + nt * 16 + l16) * 32 + 16 * half;
            acc[nt] = wmma_bf16(a.v, load_frag16(wp), acc[nt]);
        }
    }
    // C layout: row = r + 8*half, col = l16 (per 16-wide tile).
    #pragma unroll
    for (int nt = 0; nt < 4; ++nt) {
        #pragma unroll
        for (int r = 0; r < 8; ++r) {
            int row = rt * 16 + r + 8 * half;
            int col = cg * 64 + nt * 16 + l16;
            unsigned short bits = bf16_bits(acc[nt][r] + bias[col]);
            if (z == 2) {
                int bb = row >> 10, s = row & (SEQ - 1);
                int hh = col >> 6,  dd = col & (DHEAD - 1);
                OUT[(((size_t)bb * HEADS + hh) * DHEAD + dd) * SEQ + s] = bits;
            } else {
                OUT[(size_t)row * DMODEL + col] = bits;
            }
        }
    }
}

// ===========================================================================
// Kernel 4: flash-attention.  One wave owns 16 query rows of one (batch,head).
// K tiles staged to LDS with async-to-LDS transfers when available,
// context B-fragments read straight from the transposed V layout.
// ===========================================================================
__global__ __launch_bounds__(256) void attn_kernel(
        const unsigned short* __restrict__ Qb, const unsigned short* __restrict__ Kb,
        const unsigned short* __restrict__ Vt,
        const float* __restrict__ kmask, const float* __restrict__ qmask,
        unsigned short* __restrict__ ctx) {
    __shared__ unsigned short Plds[8][16 * 32];   // per-wave P tile (16x32 bf16)
#ifdef HAVE_ASYNC
    __shared__ unsigned short Klds[8][32 * 64];   // per-wave K tile, row-major
#endif

    int w = threadIdx.x >> 5, lane = threadIdx.x & 31;
    int half = lane >> 4, l16 = lane & 15;
    int wid = blockIdx.x * 8 + w;                 // 0 .. 4095
    int qt = wid & 63;
    int h  = (wid >> 6) & 15;
    int b  = wid >> 10;

    // Q fragments for head_dim = 64 -> two 16x32 A fragments, kept live.
    const unsigned short* Qrow =
        Qb + (size_t)(b * SEQ + qt * 16 + l16) * DMODEL + h * DHEAD;
    int ksel = half ? 16 : 0;
    v16bf a0, a1;
    {
        Frag16 f0, f1;
        f0.h[0] = *(const u16x8*)(Qrow + ksel);
        f0.h[1] = *(const u16x8*)(Qrow + ksel + 8);
        f1.h[0] = *(const u16x8*)(Qrow + ksel + 32);
        f1.h[1] = *(const u16x8*)(Qrow + ksel + 40);
        a0 = f0.v; a1 = f1.v;
    }

    v8f accs[4];
    #pragma unroll
    for (int nt = 0; nt < 4; ++nt)
        accs[nt] = (v8f){0.f, 0.f, 0.f, 0.f, 0.f, 0.f, 0.f, 0.f};
    float mrow[8], lrow[8];
    #pragma unroll
    for (int r = 0; r < 8; ++r) { mrow[r] = -3.0e38f; lrow[r] = 0.f; }

    const unsigned short* Kbase = Kb + (size_t)b * SEQ * DMODEL + h * DHEAD;
    const unsigned short* Vbase = Vt + ((size_t)(b * HEADS + h) * DHEAD) * SEQ;

    for (int kt = 0; kt < 32; ++kt) {
        int kbase = kt * 32;
#ifdef HAVE_ASYNC
        // Stage 32 key rows x 64 halves (4KB) with 8 async b128 copies/lane.
        wait_dscnt0();   // WAR vs previous iteration's ds reads of Klds
        #pragma unroll
        for (int i = 0; i < 8; ++i) {
            int c   = lane + 32 * i;            // 0..255 16-byte chunks
            int row = c >> 3, seg = c & 7;
            const unsigned short* g = Kbase + (size_t)(kbase + row) * DMODEL + seg * 8;
            unsigned short* l = &Klds[w][row * 64 + seg * 8];
            __builtin_amdgcn_global_load_async_to_lds_b128(
                (gv4i_t*)g, (lv4i_t*)l, 0, 0);
        }
#if __has_builtin(__builtin_amdgcn_s_wait_asynccnt)
        __builtin_amdgcn_s_wait_asynccnt(0);
#else
        asm volatile("s_wait_asynccnt 0x0" ::: "memory");
#endif
#endif
        // --- scores for 32 keys (two 16-wide WMMA N-tiles) ---
        v8f sc[2];
        #pragma unroll
        for (int t = 0; t < 2; ++t) {
            int krow = kbase + t * 16 + l16;
#ifdef HAVE_ASYNC
            const unsigned short* Ksrc = &Klds[w][(t * 16 + l16) * 64];
#else
            const unsigned short* Ksrc = Kbase + (size_t)krow * DMODEL;
            __builtin_prefetch(Ksrc + DMODEL * 32, 0, 1);
#endif
            v8f c = {0.f, 0.f, 0.f, 0.f, 0.f, 0.f, 0.f, 0.f};
            c = wmma_bf16(a0, load_frag16(Ksrc + ksel), c);
            c = wmma_bf16(a1, load_frag16(Ksrc + ksel + 32), c);
            float km = kmask[b * SEQ + krow];   // this lane's C-column == krow
            #pragma unroll
            for (int r = 0; r < 8; ++r)
                sc[t][r] = (km == 0.f) ? NEG_BIG : c[r] * 0.125f;  // 1/sqrt(64)
        }
        // --- online softmax over these 32 columns ---
        #pragma unroll
        for (int r = 0; r < 8; ++r) {
            float mx = fmaxf(sc[0][r], sc[1][r]);
            mx = fmaxf(mx, __shfl_xor(mx, 1, 32));
            mx = fmaxf(mx, __shfl_xor(mx, 2, 32));
            mx = fmaxf(mx, __shfl_xor(mx, 4, 32));
            mx = fmaxf(mx, __shfl_xor(mx, 8, 32));
            float mnew = fmaxf(mrow[r], mx);
            float corr = __expf(mrow[r] - mnew);
            float p0 = __expf(sc[0][r] - mnew);
            float p1 = __expf(sc[1][r] - mnew);
            float rs = p0 + p1;
            rs += __shfl_xor(rs, 1, 32);
            rs += __shfl_xor(rs, 2, 32);
            rs += __shfl_xor(rs, 4, 32);
            rs += __shfl_xor(rs, 8, 32);
            lrow[r] = lrow[r] * corr + rs;
            mrow[r] = mnew;
            #pragma unroll
            for (int nt = 0; nt < 4; ++nt) accs[nt][r] *= corr;
            // C-layout (row=r+8*half, col=l16) -> row-major LDS tile
            Plds[w][(r + 8 * half) * 32 + l16]      = bf16_bits(p0);
            Plds[w][(r + 8 * half) * 32 + 16 + l16] = bf16_bits(p1);
        }
        wait_dscnt0();   // same-wave LDS RAW before A-fragment reload
        // --- P as A fragment (16 x 32): contiguous 32B-aligned LDS reads ---
        v16bf pa = load_frag16(&Plds[w][l16 * 32 + ksel]);
        // --- context accumulation: acc += P(16x32) @ V(32x64) from Vt ---
        #pragma unroll
        for (int nt = 0; nt < 4; ++nt) {
            const unsigned short* vp =
                Vbase + (size_t)(nt * 16 + l16) * SEQ + kbase + 16 * half;
            accs[nt] = wmma_bf16(pa, load_frag16(vp), accs[nt]);
        }
    }
    // --- finalize: 1/l scaling and query-padding mask, store bf16 context ---
    #pragma unroll
    for (int r = 0; r < 8; ++r) {
        int row = qt * 16 + r + 8 * half;
        float scale = qmask[b * SEQ + row] / lrow[r];
        #pragma unroll
        for (int nt = 0; nt < 4; ++nt) {
            ctx[(size_t)(b * SEQ + row) * DMODEL + h * DHEAD + nt * 16 + l16] =
                bf16_bits(accs[nt][r] * scale);
        }
    }
}

// ===========================================================================
// Kernel 5: output projection: relu(ctx @ Wo + bo) + queries -> f32 scratch
// Same 16x64 register blocking as the QKV GEMM.
// ===========================================================================
__global__ __launch_bounds__(256) void out_gemm_kernel(
        const unsigned short* __restrict__ Ctx, const unsigned short* __restrict__ Wop,
        const float* __restrict__ bo, const float* __restrict__ queries,
        float* __restrict__ preln) {
    int w = threadIdx.x >> 5, lane = threadIdx.x & 31;
    int half = lane >> 4, l16 = lane & 15;
    int tile = blockIdx.x * 8 + w;          // 0 .. 4095
    int rt = tile >> 4, cg = tile & 15;
    int ksel = half ? 16 : 0;

    const unsigned short* crow = Ctx + (size_t)(rt * 16 + l16) * DMODEL;
    v8f acc[4];
    #pragma unroll
    for (int nt = 0; nt < 4; ++nt)
        acc[nt] = (v8f){0.f, 0.f, 0.f, 0.f, 0.f, 0.f, 0.f, 0.f};

    for (int kt = 0; kt < 32; ++kt) {
        Frag16 a;
        a.h[0] = *(const u16x8*)(crow + kt * 32 + ksel);
        a.h[1] = *(const u16x8*)(crow + kt * 32 + ksel + 8);
        if (kt + 1 < 32)
            __builtin_prefetch(
                Wop + ((size_t)(kt + 1) * 1024 + cg * 64 + l16) * 32, 0, 1);
        #pragma unroll
        for (int nt = 0; nt < 4; ++nt) {
            const unsigned short* wp =
                Wop + ((size_t)kt * 1024 + cg * 64 + nt * 16 + l16) * 32 + 16 * half;
            acc[nt] = wmma_bf16(a.v, load_frag16(wp), acc[nt]);
        }
    }
    #pragma unroll
    for (int nt = 0; nt < 4; ++nt) {
        #pragma unroll
        for (int r = 0; r < 8; ++r) {
            int row = rt * 16 + r + 8 * half;
            int col = cg * 64 + nt * 16 + l16;
            float v = fmaxf(acc[nt][r] + bo[col], 0.f) +
                      queries[(size_t)row * DMODEL + col];
            preln[(size_t)row * DMODEL + col] = v;
        }
    }
}

// ===========================================================================
// Kernel 6: row LayerNorm -> final output
// ===========================================================================
__global__ __launch_bounds__(256) void ln_kernel(
        const float* __restrict__ preln, const float* __restrict__ gamma,
        const float* __restrict__ beta, float* __restrict__ out) {
    __shared__ float red0[8], red1[8];
    int row = blockIdx.x, tid = threadIdx.x;
    int w = tid >> 5, lane = tid & 31;
    float s = 0.f, s2 = 0.f;
    for (int i = tid; i < DMODEL; i += 256) {
        float v = preln[(size_t)row * DMODEL + i];
        s += v; s2 += v * v;
    }
    #pragma unroll
    for (int o = 16; o; o >>= 1) { s += __shfl_xor(s, o, 32); s2 += __shfl_xor(s2, o, 32); }
    if (lane == 0) { red0[w] = s; red1[w] = s2; }
    __syncthreads();
    if (w == 0) {
        float a = (lane < 8) ? red0[lane] : 0.f;
        float c = (lane < 8) ? red1[lane] : 0.f;
        #pragma unroll
        for (int o = 4; o; o >>= 1) { a += __shfl_xor(a, o, 32); c += __shfl_xor(c, o, 32); }
        if (lane == 0) { red0[0] = a; red1[0] = c; }
    }
    __syncthreads();
    float mean = red0[0] * (1.0f / DMODEL);
    float var  = red1[0] * (1.0f / DMODEL) - mean * mean;
    float inv  = rsqrtf(var + LN_EPS);
    for (int i = tid; i < DMODEL; i += 256) {
        float v = preln[(size_t)row * DMODEL + i];
        out[(size_t)row * DMODEL + i] = (v - mean) * inv * gamma[i] + beta[i];
    }
}

// ===========================================================================
// Host-side launcher
// ===========================================================================
extern "C" void kernel_launch(void* const* d_in, const int* in_sizes, int n_in,
                              void* d_out, int out_size, void* d_ws, size_t ws_size,
                              hipStream_t stream) {
    const float* queries = (const float*)d_in[0];
    const float* keys    = (const float*)d_in[1];
    const float* values  = (const float*)d_in[2];
    const float* Wq = (const float*)d_in[3];  const float* bq = (const float*)d_in[4];
    const float* Wk = (const float*)d_in[5];  const float* bk = (const float*)d_in[6];
    const float* Wv = (const float*)d_in[7];  const float* bv = (const float*)d_in[8];
    const float* Wo = (const float*)d_in[9];  const float* bo = (const float*)d_in[10];
    const float* gamma = (const float*)d_in[11];
    const float* beta  = (const float*)d_in[12];
    float* out = (float*)d_out;

    char* ws = (char*)d_ws;
    const size_t MB = 1024ull * 1024ull;
    // Workspace layout (dead-buffer reuse: preln overlays Qb/Kb after attention)
    unsigned short* wpack = (unsigned short*)(ws + 0);          //  8 MB (4 packed weights)
    unsigned short* Qb    = (unsigned short*)(ws + 8  * MB);    //  8 MB bf16
    unsigned short* Kb    = (unsigned short*)(ws + 16 * MB);    //  8 MB bf16
    unsigned short* Vt    = (unsigned short*)(ws + 24 * MB);    //  8 MB bf16 (transposed)
    unsigned short* ctx   = (unsigned short*)(ws + 32 * MB);    //  8 MB bf16
    float*          kmask = (float*)(ws + 40 * MB);             // 16 KB
    float*          qmask = (float*)(ws + 40 * MB + 16384);     // 16 KB
    float*          preln = (float*)(ws + 8 * MB);              // 16 MB, reuses Qb+Kb

    // 1) pack all four weight matrices to bf16 WMMA-friendly layout
    pack_w_kernel<<<dim3(4096, 4), 256, 0, stream>>>(Wq, Wk, Wv, Wo, wpack);
    // 2) padding masks (one wave per row, 8192 rows total)
    masks_kernel<<<dim3(1024), 256, 0, stream>>>(queries, keys, qmask, kmask);
    // 3) Q/K/V projections (grid.y selects which; V stored transposed per head)
    qkv_gemm_kernel<<<dim3(512, 3), 256, 0, stream>>>(
        queries, keys, values, wpack, bq, bk, bv, Qb, Kb, Vt);
    // 4) flash attention: 4096 waves = 4 batches x 16 heads x 64 q-tiles
    attn_kernel<<<dim3(512), 256, 0, stream>>>(Qb, Kb, Vt, kmask, qmask, ctx);
    // 5) output projection + ReLU + residual (Wo pack is 4th slot)
    out_gemm_kernel<<<dim3(512), 256, 0, stream>>>(
        ctx, wpack + 3ull * (32u * 1024u * 32u), bo, queries, preln);
    // 6) LayerNorm
    ln_kernel<<<dim3(4096), 256, 0, stream>>>(preln, gamma, beta, out);
}
